// GATBaseline_10187662426183
// MI455X (gfx1250) — matrix operand order repaired and checked by reading, not compiled, not run on hardware
//
#include <hip/hip_runtime.h>
#include <hip/hip_bf16.h>

// ---------------------------------------------------------------------------
// GAT (2 layers, 8 heads x 16 dim) + MLP head for MI455X (gfx1250, wave32).
// GEMMs via v_wmma_f32_16x16x32_bf16; edge softmax/aggregation via f32
// atomics (L2-resident: node arrays are 25.6MB << 192MB L2).
// ---------------------------------------------------------------------------

typedef __attribute__((ext_vector_type(16))) __bf16          v16bf;
typedef __attribute__((ext_vector_type(16))) unsigned short  v16us;
typedef __attribute__((ext_vector_type(8)))  float           v8f;

#define NN   50000      // nodes
#define NE   1600000    // raw edges
#define NET  1650000    // edges + self loops
#define DIM  128
#define NH   8
#define HD   16
#define WT_STRIDE 144   // LDS row stride in halfs (multiple of 16 -> 32B aligned rows)

// float -> bf16 bits, round-to-nearest-even
__device__ __forceinline__ unsigned short f2bf(float f) {
  unsigned u = __float_as_uint(f);
  u += 0x7FFFu + ((u >> 16) & 1u);
  return (unsigned short)(u >> 16);
}

// order-preserving f32 <-> u32 encoding for atomicMax
__device__ __forceinline__ unsigned enc_f(float f) {
  unsigned u = __float_as_uint(f);
  return (u & 0x80000000u) ? ~u : (u | 0x80000000u);
}
__device__ __forceinline__ float dec_f(unsigned u) {
  return (u & 0x80000000u) ? __uint_as_float(u & 0x7FFFFFFFu)
                           : __uint_as_float(~u);
}

__device__ __forceinline__ float lrelu(float v) { return v > 0.0f ? v : 0.2f * v; }
__device__ __forceinline__ float elu(float v)   { return v > 0.0f ? v : (__expf(v) - 1.0f); }

// ---------------------------------------------------------------------------
// P[nrows x 128] = X[nrows x 128] @ W[128 x 128]   (bf16 WMMA, f32 accum)
// Block = 128 threads = 4 waves; each wave computes a 16-row x 128-col stripe.
// ---------------------------------------------------------------------------
__global__ __launch_bounds__(128) void gemm_wmma_kernel(
    const float* __restrict__ X, const float* __restrict__ W,
    float* __restrict__ P, int nrows)
{
  __shared__ alignas(32) unsigned short Wt[DIM * WT_STRIDE];  // W^T as bf16: Wt[n][k]
  const int tid = threadIdx.x;
  for (int idx = tid; idx < DIM * DIM; idx += 128) {
    const int k = idx >> 7, n = idx & 127;
    Wt[n * WT_STRIDE + k] = f2bf(W[idx]);
  }
  __syncthreads();

  const int wave  = tid >> 5;
  const int lane  = tid & 31;
  const int mbase = (blockIdx.x * 4 + wave) * 16;
  if (mbase >= nrows) return;           // whole-wave guard (EXEC stays all-1s)

  const int lrow = lane & 15;
  const int lhi  = lane >> 4;

  v8f acc[8] = {};                      // 8 head tiles of 16x16 f32
  const float* xrow = X + (size_t)(mbase + lrow) * DIM;

#pragma unroll
  for (int kt = 0; kt < 4; ++kt) {      // K = 128 in steps of 32
    const int kb = kt * 32 + lhi * 16;  // A layout: lane half h holds K = h*16 + 0..15 (matches B)
    v16us au;
#pragma unroll
    for (int i = 0; i < 8; ++i) au[i]     = f2bf(xrow[kb + i]);
#pragma unroll
    for (int i = 0; i < 8; ++i) au[8 + i] = f2bf(xrow[kb + 8 + i]);
    const v16bf a = __builtin_bit_cast(v16bf, au);

#pragma unroll
    for (int t = 0; t < 8; ++t) {       // B: lane holds col N=lane&15, K = (lane>>4)*16 + 0..15
      const v16us bu =
          *(const v16us*)&Wt[(t * 16 + lrow) * WT_STRIDE + kt * 32 + lhi * 16];
      const v16bf b = __builtin_bit_cast(v16bf, bu);
      acc[t] = __builtin_amdgcn_wmma_f32_16x16x32_bf16(
          false, a, false, b, (short)0, acc[t], false, false);
    }
  }

  // C/D layout: VGPR r -> row mbase + r + (lane>>4)*8, col = t*16 + (lane&15)
#pragma unroll
  for (int t = 0; t < 8; ++t)
#pragma unroll
    for (int r = 0; r < 8; ++r)
      P[(size_t)(mbase + r + lhi * 8) * DIM + t * 16 + lrow] = acc[t][r];
}

// alpha_src / alpha_dst: per (node, head) dot of 16-dim head slice
__global__ __launch_bounds__(256) void alpha_kernel(
    const float* __restrict__ H, const float* __restrict__ as_,
    const float* __restrict__ ad_, float* __restrict__ Asrc,
    float* __restrict__ Adst, int n)
{
  const int i = blockIdx.x * 256 + threadIdx.x;
  if (i >= n * NH) return;
  const int node = i >> 3, h = i & 7;
  const float* hp = H + (size_t)node * DIM + h * HD;
  float ss = 0.f, sd = 0.f;
#pragma unroll
  for (int c = 0; c < HD; ++c) {
    const float v = hp[c];
    ss += v * as_[h * HD + c];
    sd += v * ad_[h * HD + c];
  }
  Asrc[i] = ss;
  Adst[i] = sd;
}

__global__ __launch_bounds__(256) void init_kernel(
    float* __restrict__ out, unsigned* __restrict__ menc,
    float* __restrict__ denom, int n)
{
  const int i = blockIdx.x * 256 + threadIdx.x;
  if (i < n * DIM) out[i] = 0.0f;
  if (i < n * NH) { menc[i] = 0x007FFFFFu; denom[i] = 0.0f; }  // enc(-inf)
}

__global__ __launch_bounds__(256) void edge_max_kernel(
    const long long* __restrict__ ei, const float* __restrict__ Asrc,
    const float* __restrict__ Adst, unsigned* __restrict__ menc)
{
  const int e = blockIdx.x * 256 + threadIdx.x;
  if (e >= NET) return;
  __builtin_prefetch(&ei[e + 8192], 0, 0);        // global_prefetch_b8 (speculative)
  __builtin_prefetch(&ei[NE + e + 8192], 0, 0);
  long long s, d;
  if (e < NE) { s = ei[e]; d = ei[NE + e]; } else { s = d = e - NE; }
#pragma unroll
  for (int h = 0; h < NH; ++h) {
    const float logit = lrelu(Asrc[s * NH + h] + Adst[d * NH + h]);
    atomicMax(&menc[d * NH + h], enc_f(logit));
  }
}

__global__ __launch_bounds__(256) void edge_sum_kernel(
    const long long* __restrict__ ei, const float* __restrict__ Asrc,
    const float* __restrict__ Adst, const unsigned* __restrict__ menc,
    float* __restrict__ denom)
{
  const int e = blockIdx.x * 256 + threadIdx.x;
  if (e >= NET) return;
  __builtin_prefetch(&ei[e + 8192], 0, 0);
  __builtin_prefetch(&ei[NE + e + 8192], 0, 0);
  long long s, d;
  if (e < NE) { s = ei[e]; d = ei[NE + e]; } else { s = d = e - NE; }
#pragma unroll
  for (int h = 0; h < NH; ++h) {
    const float logit = lrelu(Asrc[s * NH + h] + Adst[d * NH + h]);
    atomicAdd(&denom[d * NH + h], __expf(logit - dec_f(menc[d * NH + h])));
  }
}

// 128 threads per edge (one per channel); scatter-add weighted messages.
__global__ __launch_bounds__(256) void edge_agg_kernel(
    const long long* __restrict__ ei, const float* __restrict__ Hsrc,
    const float* __restrict__ Asrc, const float* __restrict__ Adst,
    const unsigned* __restrict__ menc, const float* __restrict__ denom,
    float* __restrict__ out)
{
  const int gid = blockIdx.x * 256 + threadIdx.x;
  const int e = gid >> 7;
  const int c = gid & 127;
  if (e >= NET) return;
  const int h = c >> 4;
  long long s, d;
  if (e < NE) { s = ei[e]; d = ei[NE + e]; } else { s = d = e - NE; }
  const float logit = lrelu(Asrc[s * NH + h] + Adst[d * NH + h]);
  const float w = __expf(logit - dec_f(menc[d * NH + h])) / denom[d * NH + h];
  atomicAdd(&out[(size_t)d * DIM + c], Hsrc[(size_t)s * DIM + c] * w);
}

__global__ __launch_bounds__(256) void bias_elu_kernel(
    float* __restrict__ out, const float* __restrict__ bias, int n)
{
  const int i = blockIdx.x * 256 + threadIdx.x;
  if (i >= n * DIM) return;
  out[i] = elu(out[i] + bias[i & 127]);
}

__global__ __launch_bounds__(256) void cls1_kernel(
    const float* __restrict__ Hin, const float* __restrict__ cw1,
    const float* __restrict__ cb1, float* __restrict__ mid, int n)
{
  const int i = blockIdx.x * 256 + threadIdx.x;
  if (i >= n * 64) return;
  const int node = i >> 6, j = i & 63;
  const float* hp = Hin + (size_t)node * DIM;
  float sum = cb1[j];
#pragma unroll 8
  for (int k = 0; k < DIM; ++k) sum += hp[k] * cw1[k * 64 + j];
  mid[i] = elu(sum);
}

__global__ __launch_bounds__(256) void cls2_kernel(
    const float* __restrict__ mid, const float* __restrict__ cw2,
    const float* __restrict__ cb2, float* __restrict__ outp, int n)
{
  const int node = blockIdx.x * 256 + threadIdx.x;
  if (node >= n) return;
  const float* mp = mid + (size_t)node * 64;
  float l0 = cb2[0], l1 = cb2[1];
#pragma unroll 8
  for (int k = 0; k < 64; ++k) {
    const float v = mp[k];
    l0 += v * cw2[k * 2 + 0];
    l1 += v * cw2[k * 2 + 1];
  }
  outp[node * 2 + 0] = l0;
  outp[node * 2 + 1] = l1;
  const float mx = fmaxf(l0, l1);
  const float e0 = __expf(l0 - mx), e1 = __expf(l1 - mx);
  const float inv = 1.0f / (e0 + e1);
  outp[(size_t)n * 2 + node * 2 + 0] = e0 * inv;
  outp[(size_t)n * 2 + node * 2 + 1] = e1 * inv;
}

// ---------------------------------------------------------------------------
extern "C" void kernel_launch(void* const* d_in, const int* in_sizes, int n_in,
                              void* d_out, int out_size, void* d_ws, size_t ws_size,
                              hipStream_t stream)
{
  (void)in_sizes; (void)n_in; (void)out_size; (void)ws_size;

  const float*     x   = (const float*)d_in[0];
  const long long* ei  = (const long long*)d_in[1];   // int64 [2, NE]
  const float*     W1  = (const float*)d_in[2];
  const float*     as1 = (const float*)d_in[3];
  const float*     ad1 = (const float*)d_in[4];
  const float*     b1  = (const float*)d_in[5];
  const float*     W2  = (const float*)d_in[6];
  const float*     as2 = (const float*)d_in[7];
  const float*     ad2 = (const float*)d_in[8];
  const float*     b2  = (const float*)d_in[9];
  const float*     cw1 = (const float*)d_in[10];
  const float*     cb1 = (const float*)d_in[11];
  const float*     cw2 = (const float*)d_in[12];
  const float*     cb2 = (const float*)d_in[13];
  float*           outp = (float*)d_out;

  // Workspace layout (~57.6 MB total)
  char* ws = (char*)d_ws;
  float*    hA    = (float*)ws;    ws += (size_t)NN * DIM * sizeof(float);
  float*    hB    = (float*)ws;    ws += (size_t)NN * DIM * sizeof(float);
  float*    Asrc  = (float*)ws;    ws += (size_t)NN * NH * sizeof(float);
  float*    Adst  = (float*)ws;    ws += (size_t)NN * NH * sizeof(float);
  float*    denom = (float*)ws;    ws += (size_t)NN * NH * sizeof(float);
  unsigned* menc  = (unsigned*)ws; ws += (size_t)NN * NH * sizeof(unsigned);

  const int gGemm = (NN + 63) / 64;
  const int gNode = (NN * DIM + 255) / 256;
  const int gAlph = (NN * NH + 255) / 256;
  const int gEdge = (NET + 255) / 256;
  const int gAgg  = (int)(((size_t)NET * DIM + 255) / 256);

  auto layer = [&](const float* in, const float* W, const float* as_,
                   const float* ad_, const float* bias, float* hlin, float* out) {
    gemm_wmma_kernel<<<gGemm, 128, 0, stream>>>(in, W, hlin, NN);
    alpha_kernel<<<gAlph, 256, 0, stream>>>(hlin, as_, ad_, Asrc, Adst, NN);
    init_kernel<<<gNode, 256, 0, stream>>>(out, menc, denom, NN);
    edge_max_kernel<<<gEdge, 256, 0, stream>>>(ei, Asrc, Adst, menc);
    edge_sum_kernel<<<gEdge, 256, 0, stream>>>(ei, Asrc, Adst, menc, denom);
    edge_agg_kernel<<<gAgg, 256, 0, stream>>>(ei, hlin, Asrc, Adst, menc, denom, out);
    bias_elu_kernel<<<gNode, 256, 0, stream>>>(out, bias, NN);
  };

  // Layer 1: x -> hB ; Layer 2: hB -> hB (hA holds linear output / messages)
  layer(x,  W1, as1, ad1, b1, hA, hB);
  layer(hB, W2, as2, ad2, b2, hA, hB);

  // Classifier head: hB -> mid (reuse hA) -> logits + softmax
  cls1_kernel<<<(NN * 64 + 255) / 256, 256, 0, stream>>>(hB, cw1, cb1, hA, NN);
  cls2_kernel<<<(NN + 255) / 256, 256, 0, stream>>>(hA, cw2, cb2, outp, NN);
}
